// Decoder_65111704207909
// MI455X (gfx1250) — compile-verified, matrix operand order checked by reading, not run on hardware
//
#include <hip/hip_runtime.h>
#include <hip/hip_bf16.h>
#include <cstdint>
#include <cstddef>

typedef __attribute__((ext_vector_type(16))) __bf16 v16bf;
typedef __attribute__((ext_vector_type(8)))  float  v8f;

#define HD __device__ __forceinline__

// ---------------------------------------------------------------------------
// Pack U (128x512) into bf16 WMMA B-fragments (kk = 0..3 only; cond tail is
// handled in f32 VALU inside the main kernel).
// Fragment index: (jc*4 + kk)*32 + lane, each task = 16 halves (one v16bf).
//   kb = (lane>=16)?8:0 ; half h<8 -> K = kk*32 + kb + h ; else kk*32+16+kb+(h-8)
// ---------------------------------------------------------------------------
HD void pack_u(const float* U, unsigned short* out, int idx) {
    int lane = idx & 31;
    int kk   = (idx >> 5) & 3;
    int jc   = idx / 128;
    int col  = jc * 16 + (lane & 15);
    int kb   = ((lane >> 4) & 1) * 8;
#pragma unroll
    for (int h = 0; h < 16; ++h) {
        int K = kk * 32 + (h < 8 ? kb + h : 16 + kb + (h - 8));
        __bf16 b = (__bf16)U[(size_t)K * 512 + col];
        out[(size_t)idx * 16 + h] = __builtin_bit_cast(unsigned short, b);
    }
}

HD void pack_wh(const float* Wh, int ncols, unsigned short* out, int idx) {
    int lane = idx & 31;
    int kk   = (idx >> 5) & 3;
    int jc   = idx / 128;
    int col  = jc * 16 + (lane & 15);
    int kb   = ((lane >> 4) & 1) * 8;
#pragma unroll
    for (int h = 0; h < 16; ++h) {
        int K = kk * 32 + (h < 8 ? kb + h : 16 + kb + (h - 8));
        float v = (col < ncols) ? Wh[(size_t)K * ncols + col] : 0.f;
        __bf16 b = (__bf16)v;
        out[(size_t)idx * 16 + h] = __builtin_bit_cast(unsigned short, b);
    }
}

__global__ void pack_weights(const float* U_my, const float* U_ff,
                             const float* Wh_my, const float* Wh_ff,
                             unsigned short* uMyF, unsigned short* uFfF,
                             unsigned short* whMyF, unsigned short* whFfF) {
    int t = blockIdx.x * blockDim.x + threadIdx.x;
    const int nU = 32 * 4 * 32;          // 4096 tasks per U matrix
    const int nWhMy = 3 * 4 * 32;        // 384
    const int nWhFf = 2 * 4 * 32;        // 256
    if (t < nU)                          pack_u(U_my, uMyF, t);
    else if (t < 2 * nU)                 pack_u(U_ff, uFfF, t - nU);
    else if (t < 2 * nU + nWhMy)         pack_wh(Wh_my, 45, whMyF, t - 2 * nU);
    else if (t < 2 * nU + nWhMy + nWhFf) pack_wh(Wh_ff, 30, whFfF, t - 2 * nU - nWhMy);
}

// ---------------------------------------------------------------------------
// base = enc_h @ W[:128] + b, written in C-fragment order:
// idx = ((w*32 + jc)*32 + lane)*8 + v ; row = w*16 + v + 8*(lane>=16) ; col = jc*16 + lane%16
// ---------------------------------------------------------------------------
__global__ void compute_base(const float* __restrict__ enc_h, const float* __restrict__ W,
                             const float* __restrict__ bvec, float* __restrict__ outF, int B) {
    size_t idx = (size_t)blockIdx.x * blockDim.x + threadIdx.x;
    if (idx >= (size_t)B * 512) return;
    int v    = idx & 7;
    int lane = (idx >> 3) & 31;
    int jc   = (idx >> 8) & 31;
    int w    = (int)(idx >> 13);
    int row  = w * 16 + v + 8 * ((lane >> 4) & 1);
    int col  = jc * 16 + (lane & 15);
    float acc = bvec[col];
    const float* hr = enc_h + (size_t)row * 128;
    for (int k = 0; k < 128; ++k) acc += hr[k] * W[(size_t)k * 512 + col];
    outF[idx] = acc;
}

// ---------------------------------------------------------------------------
// Main persistent kernel: 128 threads (4 waves) own 16 batch rows for all T
// steps. U weights staged once into LDS via async copy; WMMA feeds from LDS.
// ---------------------------------------------------------------------------
HD float sigf(float x) { return 1.f / (1.f + __expf(-x)); }

HD void async_copy_region(void* dst_lds, const void* src_global, int bytes,
                          int tid, int nthreads) {
    uint32_t ldsbase = (uint32_t)(uintptr_t)dst_lds;   // flat-shared low 32 bits = LDS offset
    const char* g = (const char*)src_global;
    for (int off = tid * 16; off < bytes; off += nthreads * 16) {
        uint32_t l = ldsbase + (uint32_t)off;
        uint64_t a = (uint64_t)(uintptr_t)(g + off);
        asm volatile("global_load_async_to_lds_b128 %0, %1, off"
                     :: "v"(l), "v"(a) : "memory");
    }
}

HD v16bf build_a16(const __bf16* lds, int stride, int kk, int lane) {
    int rowA = lane & 15;
    int kb   = kk * 32 + ((lane >> 4) & 1) * 8;
    const __bf16* p = lds + rowA * stride + kb;
    v16bf a;
#pragma unroll
    for (int h = 0; h < 8; ++h) { a[h] = p[h]; a[h + 8] = p[16 + h]; }
    return a;
}

// One wave processes j-tiles {wave, wave+4} for all 4 gates of one LSTM.
template <int NT>
HD void lstm_tiles2(const v16bf* a, const unsigned short* uL, const float* basePtr,
                    const float (*wt)[512], float (*c)[128], __bf16 (*h)[128],
                    const float (*cpad)[8], int lane, int wave, int wblk) {
    const int hi = (lane >> 4) & 1;
    const int ns = lane & 15;
    const v16bf* uF = reinterpret_cast<const v16bf*>(uL);
#pragma unroll
    for (int jj = 0; jj < 2; ++jj) {
        int j = wave + jj * 4;
        v8f acc[4];
#pragma unroll
        for (int g = 0; g < 4; ++g) {
            int jc = g * 8 + j;
            acc[g] = *reinterpret_cast<const v8f*>(
                basePtr + (((size_t)wblk * 32 + jc) * 32 + lane) * 8);
        }
#pragma unroll
        for (int kk = 0; kk < 4; ++kk) {
#pragma unroll
            for (int g = 0; g < 4; ++g) {
                int jc = g * 8 + j;
                acc[g] = __builtin_amdgcn_wmma_f32_16x16x32_bf16(
                    false, a[kk], false, uF[(jc * 4 + kk) * 32 + lane],
                    (short)0, acc[g], false, false);
            }
        }
        int colz = j * 16 + ns;
        // cond-tail weights for this column, all 4 gates (f32, from LDS)
        float wq[NT][4];
#pragma unroll
        for (int q = 0; q < NT; ++q)
#pragma unroll
            for (int g = 0; g < 4; ++g) wq[q][g] = wt[q][g * 128 + colz];
#pragma unroll
        for (int v = 0; v < 8; ++v) {
            int row = v + 8 * hi;
            float z0 = acc[0][v], z1 = acc[1][v], z2 = acc[2][v], z3 = acc[3][v];
#pragma unroll
            for (int q = 0; q < NT; ++q) {
                float cq = cpad[row][q];
                z0 += cq * wq[q][0]; z1 += cq * wq[q][1];
                z2 += cq * wq[q][2]; z3 += cq * wq[q][3];
            }
            float iv = sigf(z0), fv = sigf(z1), gv = tanhf(z2), ov = sigf(z3);
            float c2 = fv * c[row][colz] + iv * gv;
            c[row][colz] = c2;
            h[row][colz] = (__bf16)(ov * tanhf(c2));
        }
    }
}

HD void head_tile(const v16bf* a, const v16bf* whF, const float* bh,
                  int ncols, int jc, float* rout, int rstride, int lane) {
    const int hi = (lane >> 4) & 1;
    const int ns = lane & 15;
    v8f acc;
#pragma unroll
    for (int v = 0; v < 8; ++v) acc[v] = 0.f;
#pragma unroll
    for (int kk = 0; kk < 4; ++kk)
        acc = __builtin_amdgcn_wmma_f32_16x16x32_bf16(
            false, a[kk], false, whF[(jc * 4 + kk) * 32 + lane],
            (short)0, acc, false, false);
    int col = jc * 16 + ns;
    float bias = (col < ncols) ? bh[col] : 0.f;
#pragma unroll
    for (int v = 0; v < 8; ++v) rout[(v + 8 * hi) * rstride + col] = acc[v] + bias;
}

HD void softmax5(const float* lg, float* out) {
    float m = lg[0];
#pragma unroll
    for (int k = 1; k < 5; ++k) m = fmaxf(m, lg[k]);
    float s = 0.f;
#pragma unroll
    for (int k = 0; k < 5; ++k) { out[k] = __expf(lg[k] - m); s += out[k]; }
    float inv = 1.f / s;
#pragma unroll
    for (int k = 0; k < 5; ++k) out[k] *= inv;
}

HD int argmax5g(const float* lg, const float* g) {
    float bv = lg[0] + g[0]; int bi = 0;
#pragma unroll
    for (int k = 1; k < 5; ++k) { float v = lg[k] + g[k]; if (v > bv) { bv = v; bi = k; } }
    return bi;
}

HD float pick5(const float* a, int i) {
    float v = a[0];
#pragma unroll
    for (int k = 1; k < 5; ++k) v = (k == i) ? a[k] : v;
    return v;
}

__global__ __launch_bounds__(128)
void decoder_main(const float* __restrict__ cond_m, const float* __restrict__ cond_y,
                  const float* __restrict__ cond_f, const float* __restrict__ cond_fa,
                  const float* __restrict__ state_h, const float* __restrict__ state_c,
                  const float* __restrict__ W_my, const float* __restrict__ W_ff,
                  const float* __restrict__ bh_my, const float* __restrict__ bh_ff,
                  const float* __restrict__ gumbel, const float* __restrict__ znorm,
                  const float* __restrict__ baseMy, const float* __restrict__ baseFf,
                  const unsigned short* __restrict__ uMyP, const unsigned short* __restrict__ uFfP,
                  const unsigned short* __restrict__ whMyP, const unsigned short* __restrict__ whFfP,
                  float* __restrict__ gm, float* __restrict__ gy,
                  float* __restrict__ gf, float* __restrict__ gfa,
                  int B, int T) {
    // ---- LDS: ~307 KB total (fits the 320 KB per-workgroup limit) ----
    __shared__ __align__(32) unsigned short uMyL[32 * 4 * 32 * 16];  // 128 KB
    __shared__ __align__(32) unsigned short uFfL[32 * 4 * 32 * 16];  // 128 KB
    __shared__ float wtm[5][512];                                    // 10 KB  (W_my cond tail)
    __shared__ float wtf[2][512];                                    //  4 KB  (W_ff cond tail)
    __shared__ float c_my[16][128], c_ff[16][128];                   // 16 KB
    __shared__ __align__(16) __bf16 h_my[16][128], h_ff[16][128];    //  8 KB
    __shared__ float cpad_my[16][8], cpad_ff[16][8];                 //  1 KB
    __shared__ float r_my[16][48], r_ff[16][32];                     //  5 KB

    const int tid  = threadIdx.x;
    const int lane = tid & 31;
    const int wave = tid >> 5;
    const int wblk = blockIdx.x;
    const int rowbase = wblk * 16;

    // ---- async-stage step-invariant weights into LDS (reused T times) ----
    async_copy_region(uMyL, uMyP, 32 * 4 * 32 * 32, tid, 128);
    async_copy_region(uFfL, uFfP, 32 * 4 * 32 * 32, tid, 128);
    async_copy_region(&wtm[0][0], W_my + 128 * 512, 5 * 512 * 4, tid, 128);
    async_copy_region(&wtf[0][0], W_ff + 128 * 512, 2 * 512 * 4, tid, 128);

    // ---- init state into LDS (overlaps with async copies) ----
    for (int i = tid; i < 16 * 128; i += 128) {
        int r = i >> 7, k = i & 127;
        float hv = state_h[(size_t)(rowbase + r) * 128 + k];
        float cv = state_c[(size_t)(rowbase + r) * 128 + k];
        c_my[r][k] = cv; c_ff[r][k] = cv;
        h_my[r][k] = (__bf16)hv; h_ff[r][k] = (__bf16)hv;
    }
    if (tid < 16 * 8) { int r = tid >> 3, k = tid & 7; cpad_my[r][k] = 0.f; cpad_ff[r][k] = 0.f; }
    if (tid < 16) {
        int b = rowbase + tid;
        float m0  = cond_m[((size_t)b * T) * 2 + 0];
        float m1  = cond_m[((size_t)b * T) * 2 + 1];
        float y0  = cond_y[(size_t)b * T];
        float f0  = cond_f[(size_t)b * T];
        float fa0 = cond_fa[(size_t)b * T];
        cpad_my[tid][0] = m0; cpad_my[tid][1] = m1; cpad_my[tid][2] = y0;
        cpad_my[tid][3] = f0; cpad_my[tid][4] = fa0;
        cpad_ff[tid][0] = f0; cpad_ff[tid][1] = fa0;
    }
    asm volatile("s_wait_asynccnt 0" ::: "memory");
    __syncthreads();

    const v16bf* whMyF = reinterpret_cast<const v16bf*>(whMyP);
    const v16bf* whFfF = reinterpret_cast<const v16bf*>(whFfP);

    for (int t = 0; t < T; ++t) {
        // ---- recurrent GEMMs (A rebuilt per phase; nothing live across phases) ----
        {
            v16bf a[4];
#pragma unroll
            for (int kk = 0; kk < 4; ++kk) a[kk] = build_a16(&h_my[0][0], 128, kk, lane);
            lstm_tiles2<5>(a, uMyL, baseMy, wtm, c_my, h_my, cpad_my, lane, wave, wblk);
        }
        {
            v16bf a[4];
#pragma unroll
            for (int kk = 0; kk < 4; ++kk) a[kk] = build_a16(&h_ff[0][0], 128, kk, lane);
            lstm_tiles2<2>(a, uFfL, baseFf, wtf, c_ff, h_ff, cpad_ff, lane, wave, wblk);
        }
        __syncthreads();   // h complete before head A rebuild

        // ---- heads: waves 0..2 -> r_my tiles; wave 3 -> both r_ff tiles ----
        if (wave < 3) {
            v16bf a[4];
#pragma unroll
            for (int kk = 0; kk < 4; ++kk) a[kk] = build_a16(&h_my[0][0], 128, kk, lane);
            head_tile(a, whMyF, bh_my, 45, wave, &r_my[0][0], 48, lane);
        } else {
            v16bf a[4];
#pragma unroll
            for (int kk = 0; kk < 4; ++kk) a[kk] = build_a16(&h_ff[0][0], 128, kk, lane);
            head_tile(a, whFfF, bh_ff, 30, 0, &r_ff[0][0], 32, lane);
            head_tile(a, whFfF, bh_ff, 30, 1, &r_ff[0][0], 32, lane);
        }
        __syncthreads();   // r complete before sampling

        // ---- per-row mixture / sampling / cond update (wave 0, lanes 0..15) ----
        if (wave == 0 && lane < 16) {
            int r = lane, b = rowbase + r;
            float rm[45], rf[30];
#pragma unroll
            for (int k = 0; k < 45; ++k) rm[k] = r_my[r][k];
#pragma unroll
            for (int k = 0; k < 30; ++k) rf[k] = r_ff[r][k];

            float am[5], ay[5], af[5], afa[5];
            softmax5(rm + 0,  am);
            softmax5(rm + 30, ay);
            softmax5(rf + 0,  af);
            softmax5(rf + 15, afa);

            float sl[5], slat[5], rho[5], sy[5], sf[5], sfa[5];
#pragma unroll
            for (int k = 0; k < 5; ++k) {
                sl[k]   = __expf(rm[10 + k]);
                slat[k] = __expf(rm[20 + k]);
                rho[k]  = tanhf(rm[25 + k]);
                sy[k]   = __expf(rm[40 + k]);
                sf[k]   = __expf(rf[10 + k]);
                sfa[k]  = __expf(rf[25 + k]);
            }

            size_t ob = (size_t)b * T + t;
            float* pm  = gm  + ob * 30;
            float* py  = gy  + ob * 15;
            float* pf  = gf  + ob * 15;
            float* pfa = gfa + ob * 15;
#pragma unroll
            for (int k = 0; k < 5; ++k) {
                pm[k] = am[k];           pm[5 + k] = rm[5 + k];   pm[10 + k] = sl[k];
                pm[15 + k] = rm[15 + k]; pm[20 + k] = slat[k];    pm[25 + k] = rho[k];
                py[k] = ay[k];           py[5 + k] = rm[35 + k];  py[10 + k] = sy[k];
                pf[k] = af[k];           pf[5 + k] = rf[5 + k];   pf[10 + k] = sf[k];
                pfa[k] = afa[k];         pfa[5 + k] = rf[20 + k]; pfa[10 + k] = sfa[k];
            }

            float g0[5], g1[5], g2[5], g3[5], zn[5];
#pragma unroll
            for (int k = 0; k < 5; ++k) {
                g0[k] = gumbel[(((size_t)t * 4 + 0) * B + b) * 5 + k];
                g1[k] = gumbel[(((size_t)t * 4 + 1) * B + b) * 5 + k];
                g2[k] = gumbel[(((size_t)t * 4 + 2) * B + b) * 5 + k];
                g3[k] = gumbel[(((size_t)t * 4 + 3) * B + b) * 5 + k];
                zn[k] = znorm[((size_t)t * B + b) * 5 + k];
            }
            int im = argmax5g(rm + 0, g0);
            float z1 = zn[0], z2 = zn[1];
            float rr     = pick5(rho, im);
            float s_long = pick5(rm + 5, im)  + pick5(sl, im)   * z1;
            float s_lt   = pick5(rm + 15, im) + pick5(slat, im) *
                           (rr * z1 + sqrtf(fmaxf(0.f, 1.f - rr * rr)) * z2);
            int iy = argmax5g(rm + 30, g1);
            float samp_y = pick5(rm + 35, iy) + pick5(sy, iy) * zn[2];
            int ifx = argmax5g(rf + 0, g2);
            float samp_f = pick5(rf + 5, ifx) + pick5(sf, ifx) * zn[3];
            int ia = argmax5g(rf + 15, g3);
            float samp_fa = pick5(rf + 20, ia) + pick5(sfa, ia) * zn[4];

            int tn = (t + 1 < T) ? (t + 1) : (T - 1);
            float nm0  = cond_m[((size_t)b * T + tn) * 2 + 0];
            float nm1  = cond_m[((size_t)b * T + tn) * 2 + 1];
            float ny   = cond_y[(size_t)b * T + tn];
            float nf_  = cond_f[(size_t)b * T + tn];
            float nfa_ = cond_fa[(size_t)b * T + tn];

            float cm0  = (fabsf(s_long - nm0) < 0.3f) ? s_long : nm0;
            float cm1  = (fabsf(s_lt  - nm1) < 0.1f) ? s_lt  : nm1;
            float cy   = (fabsf(samp_y - ny)  < 0.3f) ? samp_y : ny;
            float cf_  = (fabsf(samp_f - nf_) < 0.3f) ? samp_f : nf_;
            float cfa_ = (fabsf(samp_fa - nfa_) < 0.3f) ? samp_fa : nfa_;

            cpad_my[r][0] = cm0; cpad_my[r][1] = cm1; cpad_my[r][2] = cy;
            cpad_my[r][3] = cf_; cpad_my[r][4] = cfa_;
            cpad_ff[r][0] = cf_; cpad_ff[r][1] = cfa_;
        }
        __syncthreads();   // cpad visible to all waves for next step
    }
}

// ---------------------------------------------------------------------------
extern "C" void kernel_launch(void* const* d_in, const int* in_sizes, int n_in,
                              void* d_out, int out_size, void* d_ws, size_t ws_size,
                              hipStream_t stream) {
    (void)n_in; (void)out_size; (void)ws_size;
    const float* cond_m  = (const float*)d_in[0];
    const float* cond_y  = (const float*)d_in[1];
    const float* cond_f  = (const float*)d_in[2];
    const float* cond_fa = (const float*)d_in[3];
    const float* state_h = (const float*)d_in[4];
    const float* state_c = (const float*)d_in[5];
    const float* W_my    = (const float*)d_in[6];
    const float* U_my    = (const float*)d_in[7];
    const float* b_my    = (const float*)d_in[8];
    const float* W_ff    = (const float*)d_in[9];
    const float* U_ff    = (const float*)d_in[10];
    const float* b_ff    = (const float*)d_in[11];
    const float* Wh_my   = (const float*)d_in[12];
    const float* bh_my   = (const float*)d_in[13];
    const float* Wh_ff   = (const float*)d_in[14];
    const float* bh_ff   = (const float*)d_in[15];
    const float* gumbel  = (const float*)d_in[16];
    const float* znorm   = (const float*)d_in[17];

    const int H = 128;
    const int B = in_sizes[4] / H;          // 8192
    const int T = in_sizes[1] / B;          // 20

    char* ws = (char*)d_ws;
    size_t szBase = (size_t)B * 512 * sizeof(float);
    float* baseMy = (float*)(ws);
    float* baseFf = (float*)(ws + szBase);
    size_t szUF   = (size_t)32 * 4 * 32 * 16 * sizeof(unsigned short);  // 128 KB
    size_t szWhMy = (size_t)3 * 4 * 32 * 16 * sizeof(unsigned short);
    unsigned short* uMyF  = (unsigned short*)(ws + 2 * szBase);
    unsigned short* uFfF  = (unsigned short*)((char*)uMyF + szUF);
    unsigned short* whMyF = (unsigned short*)((char*)uFfF + szUF);
    unsigned short* whFfF = (unsigned short*)((char*)whMyF + szWhMy);

    int packTasks = 2 * (32 * 4 * 32) + 3 * 4 * 32 + 2 * 4 * 32;
    pack_weights<<<(packTasks + 255) / 256, 256, 0, stream>>>(
        U_my, U_ff, Wh_my, Wh_ff, uMyF, uFfF, whMyF, whFfF);

    size_t nbase = (size_t)B * 512;
    int nb = (int)((nbase + 255) / 256);
    compute_base<<<nb, 256, 0, stream>>>(state_h, W_my, b_my, baseMy, B);
    compute_base<<<nb, 256, 0, stream>>>(state_h, W_ff, b_ff, baseFf, B);

    float* gm  = (float*)d_out;
    float* gy  = gm + (size_t)B * T * 30;
    float* gf  = gy + (size_t)B * T * 15;
    float* gfa = gf + (size_t)B * T * 15;

    decoder_main<<<B / 16, 128, 0, stream>>>(
        cond_m, cond_y, cond_f, cond_fa, state_h, state_c, W_my, W_ff,
        bh_my, bh_ff, gumbel, znorm, baseMy, baseFf, uMyF, uFfF, whMyF, whFfF,
        gm, gy, gf, gfa, B, T);
}
